// LowRankSSMScan_87625922773418
// MI455X (gfx1250) — compile-verified, matrix operand order checked by reading, not run on hardware
//
#include <hip/hip_runtime.h>
#include <hip/hip_bf16.h>

// ---------------------------------------------------------------------------
// Problem constants (reference: B=4, L=4096, D=1024, S=64)
// ---------------------------------------------------------------------------
#define BSZ   4
#define SEQ   4096
#define DM    1024
#define ST    64
#define MROWS (BSZ * SEQ)            // 16384 rows (b,l flattened)
#define CHUNK 128                    // scan chunk (divides SEQ)
#define NCHUNK_TOT (MROWS / CHUNK)   // 128 chunks, 32 per batch

typedef __attribute__((ext_vector_type(16))) __bf16 v16bf;
typedef __attribute__((ext_vector_type(8)))  __bf16 v8bf;
typedef __attribute__((ext_vector_type(8)))  float  v8f;

// ---------------------------------------------------------------------------
// WMMA helpers: v_wmma_f32_16x16x32_bf16, D = A(16x32) x B(32x16) + C
// A: lane lo=M, hi selects K-halves; elem j: K = (j<8 ? hi*8+j : 16+hi*8+j-8)
// B: lane lo=N, same K mapping. C/D: elem i -> M = i + 8*hi, N = lo
// ---------------------------------------------------------------------------
__device__ __forceinline__ v16bf frag_ld(const __bf16* rowBase, int kBase, int hi) {
  const v8bf* p0 = reinterpret_cast<const v8bf*>(rowBase + kBase + hi * 8);
  const v8bf* p1 = reinterpret_cast<const v8bf*>(rowBase + kBase + 16 + hi * 8);
  v8bf a0 = *p0;
  v8bf a1 = *p1;
  return __builtin_shufflevector(a0, a1, 0,1,2,3,4,5,6,7,8,9,10,11,12,13,14,15);
}

__device__ __forceinline__ v8f wmma_bf16(v16bf a, v16bf b, v8f c) {
  return __builtin_amdgcn_wmma_f32_16x16x32_bf16(false, a, false, b, (short)0, c,
                                                 false, false);
}

// Async DMA: 16B per lane, global -> LDS, tracked by ASYNCcnt (CDNA5).
__device__ __forceinline__ void async_b128(uint32_t lds_off, const void* gptr) {
  asm volatile("global_load_async_to_lds_b128 %0, %1, off"
               :: "v"(lds_off), "v"((uint64_t)(uintptr_t)gptr) : "memory");
}
__device__ __forceinline__ void wait_async0() {
  asm volatile("s_wait_asynccnt 0x0" ::: "memory");
}
__device__ __forceinline__ uint32_t lds_addr(const void* p) {
  return (uint32_t)(uintptr_t)p;   // LDS aperture: addr[31:0] == LDS byte offset
}

// ---------------------------------------------------------------------------
// Kernel 0: f32 -> bf16 conversion (one pass; enables pure-DMA GEMM staging,
// halves per-GEMM read bytes; x is consumed twice downstream)
// ---------------------------------------------------------------------------
__global__ __launch_bounds__(256) void k_cvt(const float* __restrict__ src,
                                             __bf16* __restrict__ dst, int n4) {
  int i = blockIdx.x * 256 + threadIdx.x;
  if (i < n4) {
    float4 v = reinterpret_cast<const float4*>(src)[i];
    __bf16* d = dst + (size_t)i * 4;
    d[0] = (__bf16)v.x; d[1] = (__bf16)v.y; d[2] = (__bf16)v.z; d[3] = (__bf16)v.w;
  }
}

// ---------------------------------------------------------------------------
// Kernel 1: Bu_bf[r,s] = bf16( x @ B_w^T )  (M=16384, N=64, K=1024)
// 256 thr (8 waves), tile 128x64, double-buffered async LDS staging
// ---------------------------------------------------------------------------
#define K1_LDA 40   // 32 K + 16B pad (16B-aligned rows, bank-conflict break)
__global__ __launch_bounds__(256) void k_bu(const __bf16* __restrict__ xb,
                                            const __bf16* __restrict__ Bwb,
                                            __bf16* __restrict__ BuBf) {
  __shared__ __align__(16) __bf16 As[2][128 * K1_LDA];
  __shared__ __align__(16) __bf16 Bs[2][64 * K1_LDA];
  const int tid = threadIdx.x, wave = tid >> 5, lane = tid & 31;
  const int lo = lane & 15, hi = lane >> 4;
  const int row0 = blockIdx.x * 128;
  const uint32_t asB[2] = { lds_addr(&As[0][0]), lds_addr(&As[1][0]) };
  const uint32_t bsB[2] = { lds_addr(&Bs[0][0]), lds_addr(&Bs[1][0]) };

  // fixed per-thread copy slots (16B chunks); only k0 varies per iteration
  const int rA0 = tid >> 2,          cA = tid & 3;     // A chunk 0: rows 0..63
  const int rA1 = (256 + tid) >> 2;                    // A chunk 1: rows 64..127
  const __bf16* gA0 = xb + (size_t)(row0 + rA0) * DM + cA * 8;
  const __bf16* gA1 = xb + (size_t)(row0 + rA1) * DM + cA * 8;
  const __bf16* gB0 = Bwb + (size_t)rA0 * DM + cA * 8; // B: 64 rows, 1 chunk/thr
  const uint32_t lA0 = (uint32_t)((rA0 * K1_LDA + cA * 8) * 2);
  const uint32_t lA1 = (uint32_t)((rA1 * K1_LDA + cA * 8) * 2);
  const uint32_t lB0 = (uint32_t)((rA0 * K1_LDA + cA * 8) * 2);

  auto stage = [&](int k0, int b) {
    async_b128(asB[b] + lA0, gA0 + k0);
    async_b128(asB[b] + lA1, gA1 + k0);
    async_b128(bsB[b] + lB0, gB0 + k0);
  };

  v8f acc[4] = {};
  stage(0, 0);
  #pragma unroll 2     // constant buffer index; loop-carried addresses stay small
  for (int kc = 0; kc < DM / 32; ++kc) {
    int b = kc & 1;
    wait_async0();        // own stage into buf b complete
    __syncthreads();      // everyone's stage done; buf b^1 no longer being read
    if (kc + 1 < DM / 32) stage((kc + 1) * 32, b ^ 1);   // DMA overlaps compute
    v16bf af = frag_ld(&As[b][(wave * 16 + lo) * K1_LDA], 0, hi);
    #pragma unroll
    for (int nt = 0; nt < 4; ++nt) {
      v16bf bf = frag_ld(&Bs[b][(nt * 16 + lo) * K1_LDA], 0, hi);
      acc[nt] = wmma_bf16(af, bf, acc[nt]);
    }
  }

  #pragma unroll
  for (int nt = 0; nt < 4; ++nt)
    #pragma unroll
    for (int i = 0; i < 8; ++i) {
      int r = row0 + wave * 16 + hi * 8 + i;
      int c = nt * 16 + lo;
      BuBf[(size_t)r * ST + c] = (__bf16)acc[nt][i];
    }
}

// ---------------------------------------------------------------------------
// Kernel 2: u[r,n] = decay_n^l * (Bu @ C_w^T)[r,n],  l = r % SEQ
// (M=16384, N=1024, K=64); tile 128x128; single async stage (K fits)
// ---------------------------------------------------------------------------
#define K2_LD 72   // 64 K + pad
__global__ __launch_bounds__(256) void k_buproj(const __bf16* __restrict__ BuBf,
                                                const __bf16* __restrict__ Cwb,
                                                const float* __restrict__ log_decay,
                                                float* __restrict__ u) {
  __shared__ __align__(16) __bf16 As[128 * K2_LD];
  __shared__ __align__(16) __bf16 Bs[128 * K2_LD];
  const int tid = threadIdx.x, wave = tid >> 5, lane = tid & 31;
  const int lo = lane & 15, hi = lane >> 4;
  const int row0 = blockIdx.x * 128, n0 = blockIdx.y * 128;
  const uint32_t asB = lds_addr(&As[0]), bsB = lds_addr(&Bs[0]);

  #pragma unroll
  for (int i = 0; i < 4; ++i) {            // A: 128x64 bf16 = 1024 chunks
    int idx = i * 256 + tid;
    int r = idx >> 3, c = idx & 7;
    async_b128(asB + (uint32_t)((r * K2_LD + c * 8) * 2),
               BuBf + (size_t)(row0 + r) * ST + c * 8);
    async_b128(bsB + (uint32_t)((r * K2_LD + c * 8) * 2),   // B: C_w rows n0..+127
               Cwb + (size_t)(n0 + r) * ST + c * 8);
  }
  wait_async0();
  __syncthreads();

  v8f acc[8] = {};
  #pragma unroll
  for (int kc = 0; kc < 2; ++kc) {
    v16bf af = frag_ld(&As[(wave * 16 + lo) * K2_LD], kc * 32, hi);
    #pragma unroll
    for (int nt = 0; nt < 8; ++nt) {
      v16bf bf = frag_ld(&Bs[(nt * 16 + lo) * K2_LD], kc * 32, hi);
      acc[nt] = wmma_bf16(af, bf, acc[nt]);
    }
  }

  // fold decay^l into u so the scan becomes a plain exclusive cumsum
  #pragma unroll
  for (int nt = 0; nt < 8; ++nt) {
    int n = n0 + nt * 16 + lo;
    float z = log_decay[n];
    float dec = 1.0f / (1.0f + __expf(-z));  // sigmoid
    float l2d = __log2f(dec);
    #pragma unroll
    for (int i = 0; i < 8; ++i) {
      int r = row0 + wave * 16 + hi * 8 + i;
      int l = r & (SEQ - 1);
      float w = exp2f((float)l * l2d);       // decay^l (underflow -> 0, as in ref)
      u[(size_t)r * DM + n] = w * acc[nt][i];
    }
  }
}

// ---------------------------------------------------------------------------
// Scan phase 1: per-chunk sums.  grid (128 chunks, DM/256)
// ---------------------------------------------------------------------------
__global__ __launch_bounds__(256) void k_chunksum(const float* __restrict__ u,
                                                  float* __restrict__ partial) {
  const int d = blockIdx.y * 256 + threadIdx.x;
  const int cg = blockIdx.x;
  const float* p = u + (size_t)cg * CHUNK * DM + d;
  float acc = 0.f;
  #pragma unroll 4
  for (int i = 0; i < CHUNK; ++i) acc += p[(size_t)i * DM];
  partial[(size_t)cg * DM + d] = acc;
}

// ---------------------------------------------------------------------------
// Scan phase 2: exclusive scan of 32 chunk sums per (b,d), in place
// ---------------------------------------------------------------------------
__global__ __launch_bounds__(256) void k_chunkscan(float* __restrict__ partial) {
  const int idx = blockIdx.x * 256 + threadIdx.x;  // 0..4095 = (b,d)
  const int b = idx >> 10, d = idx & (DM - 1);
  float run = 0.f;
  for (int c = 0; c < SEQ / CHUNK; ++c) {
    size_t o = ((size_t)(b * (SEQ / CHUNK) + c)) * DM + d;
    float t = partial[o];
    partial[o] = run;
    run += t;
  }
}

// ---------------------------------------------------------------------------
// Scan phase 3: h[r,d] = offset + in-chunk exclusive prefix (in place on u)
// ---------------------------------------------------------------------------
__global__ __launch_bounds__(256) void k_applyscan(float* __restrict__ u,
                                                   const float* __restrict__ partial) {
  const int d = blockIdx.y * 256 + threadIdx.x;
  const int cg = blockIdx.x;
  float acc = partial[(size_t)cg * DM + d];
  float* p = u + (size_t)cg * CHUNK * DM + d;
  for (int i = 0; i < CHUNK; ++i) {
    float v = p[(size_t)i * DM];
    p[(size_t)i * DM] = acc;
    acc += v;
  }
}

// ---------------------------------------------------------------------------
// Kernel 6: out = sigmoid(x @ gate_w^T + b) * h  (M=16384, N=1024, K=1024)
// tile 128x128, double-buffered async LDS staging
// ---------------------------------------------------------------------------
#define K6_LD 40
__global__ __launch_bounds__(256) void k_gate(const __bf16* __restrict__ xb,
                                              const __bf16* __restrict__ Gwb,
                                              const float* __restrict__ gb,
                                              const float* __restrict__ h,
                                              float* __restrict__ out) {
  __shared__ __align__(16) __bf16 As[2][128 * K6_LD];
  __shared__ __align__(16) __bf16 Bs[2][128 * K6_LD];
  const int tid = threadIdx.x, wave = tid >> 5, lane = tid & 31;
  const int lo = lane & 15, hi = lane >> 4;
  const int row0 = blockIdx.x * 128, n0 = blockIdx.y * 128;
  const uint32_t asB[2] = { lds_addr(&As[0][0]), lds_addr(&As[1][0]) };
  const uint32_t bsB[2] = { lds_addr(&Bs[0][0]), lds_addr(&Bs[1][0]) };

  // fixed per-thread copy slots; only k0 varies
  const int r0 = tid >> 2,           c0 = tid & 3;   // chunk 0: rows 0..63
  const int r1 = (256 + tid) >> 2;                   // chunk 1: rows 64..127
  const __bf16* gA0 = xb  + (size_t)(row0 + r0) * DM + c0 * 8;
  const __bf16* gA1 = xb  + (size_t)(row0 + r1) * DM + c0 * 8;
  const __bf16* gB0 = Gwb + (size_t)(n0 + r0) * DM + c0 * 8;
  const __bf16* gB1 = Gwb + (size_t)(n0 + r1) * DM + c0 * 8;
  const uint32_t l0 = (uint32_t)((r0 * K6_LD + c0 * 8) * 2);
  const uint32_t l1 = (uint32_t)((r1 * K6_LD + c0 * 8) * 2);

  auto stage = [&](int k0, int b) {
    async_b128(asB[b] + l0, gA0 + k0);
    async_b128(asB[b] + l1, gA1 + k0);
    async_b128(bsB[b] + l0, gB0 + k0);
    async_b128(bsB[b] + l1, gB1 + k0);
  };

  v8f acc[8] = {};
  stage(0, 0);
  #pragma unroll 2     // constant buffer index; small loop-carried address state
  for (int kc = 0; kc < DM / 32; ++kc) {
    int b = kc & 1;
    wait_async0();
    __syncthreads();
    if (kc + 1 < DM / 32) stage((kc + 1) * 32, b ^ 1);
    v16bf af = frag_ld(&As[b][(wave * 16 + lo) * K6_LD], 0, hi);
    #pragma unroll
    for (int nt = 0; nt < 8; ++nt) {
      v16bf bf = frag_ld(&Bs[b][(nt * 16 + lo) * K6_LD], 0, hi);
      acc[nt] = wmma_bf16(af, bf, acc[nt]);
    }
  }

  #pragma unroll
  for (int nt = 0; nt < 8; ++nt) {
    int n = n0 + nt * 16 + lo;
    float bias = gb[n];
    #pragma unroll
    for (int i = 0; i < 8; ++i) {
      int r = row0 + wave * 16 + hi * 8 + i;
      float logit = acc[nt][i] + bias;
      float g = 1.0f / (1.0f + __expf(-logit));
      out[(size_t)r * DM + n] = g * h[(size_t)r * DM + n];
    }
  }
}

// ---------------------------------------------------------------------------
// Launch
// ---------------------------------------------------------------------------
extern "C" void kernel_launch(void* const* d_in, const int* in_sizes, int n_in,
                              void* d_out, int out_size, void* d_ws, size_t ws_size,
                              hipStream_t stream) {
  (void)in_sizes; (void)n_in; (void)out_size; (void)ws_size;
  const float* x  = (const float*)d_in[0];   // (B, L, D)
  const float* Bw = (const float*)d_in[1];   // (S, D)
  const float* Cw = (const float*)d_in[2];   // (D, S)
  const float* Gw = (const float*)d_in[3];   // (D, D)
  const float* gb = (const float*)d_in[4];   // (D,)
  const float* ld = (const float*)d_in[5];   // (D,)
  float* out = (float*)d_out;

  char* ws = (char*)d_ws;
  size_t off = 0;
  float*  u       = (float*)(ws + off);  off += (size_t)MROWS * DM * 4;  // 64 MB
  __bf16* xb      = (__bf16*)(ws + off); off += (size_t)MROWS * DM * 2;  // 32 MB
  __bf16* Gwb     = (__bf16*)(ws + off); off += (size_t)DM * DM * 2;     //  2 MB
  __bf16* BuBf    = (__bf16*)(ws + off); off += (size_t)MROWS * ST * 2;  //  2 MB
  __bf16* Bwb     = (__bf16*)(ws + off); off += (size_t)ST * DM * 2;
  __bf16* Cwb     = (__bf16*)(ws + off); off += (size_t)DM * ST * 2;
  float*  partial = (float*)(ws + off);

  // f32 -> bf16 conversions (enable async-DMA staging in the GEMMs)
  k_cvt<<<dim3((MROWS * DM / 4) / 256), 256, 0, stream>>>(x,  xb,  MROWS * DM / 4);
  k_cvt<<<dim3((DM * DM / 4) / 256),    256, 0, stream>>>(Gw, Gwb, DM * DM / 4);
  k_cvt<<<dim3((ST * DM / 4) / 256),    256, 0, stream>>>(Bw, Bwb, ST * DM / 4);
  k_cvt<<<dim3((DM * ST / 4) / 256),    256, 0, stream>>>(Cw, Cwb, DM * ST / 4);

  k_bu       <<<dim3(MROWS / 128),           256, 0, stream>>>(xb, Bwb, BuBf);
  k_buproj   <<<dim3(MROWS / 128, DM / 128), 256, 0, stream>>>(BuBf, Cwb, ld, u);
  k_chunksum <<<dim3(NCHUNK_TOT, DM / 256),  256, 0, stream>>>(u, partial);
  k_chunkscan<<<dim3((BSZ * DM) / 256),      256, 0, stream>>>(partial);
  k_applyscan<<<dim3(NCHUNK_TOT, DM / 256),  256, 0, stream>>>(u, partial);
  k_gate     <<<dim3(MROWS / 128, DM / 128), 256, 0, stream>>>(xb, Gwb, gb, u, out);
}
